// d128snn_delays_2628519985330
// MI455X (gfx1250) — matrix operand order repaired and checked by reading, not compiled, not run on hardware
//
#include <hip/hip_runtime.h>

// ---------------------------------------------------------------------------
// MI455X (gfx1250) DCLS-delay SNN forward pass.
// ~160 GFLOP of conv math on v_wmma_f32_16x16x32_bf16 (bf16 in, f32 acc).
// Layers 1/2: LDS-staged block GEMM, one workgroup per batch. The 136-row
// padded time column of each 32-wide ci chunk is staged once and reused by
// all 25 taps; per tap a 32x256 kernel slice is streamed into a DOUBLE-
// BUFFERED LDS tile via gfx1250 async global->LDS loads (ASYNCcnt) while
// the current tap's 112 WMMAs execute -> one barrier per tap, ~150 B of L2
// traffic per 16.4 KFLOP WMMA.
// ---------------------------------------------------------------------------

typedef __bf16 bf16_t;
typedef __bf16 v16bf __attribute__((ext_vector_type(16)));
typedef float  v8f   __attribute__((ext_vector_type(8)));
typedef unsigned int uiv4 __attribute__((ext_vector_type(4)));

union BFrag { v16bf v; uiv4 q[2]; };

#define BATCH 128
#define TREAL 100
#define TPADO 112          // output time padded to 7*16
#define TROWS 136          // 24 causal left-pad rows + 112
#define LTAPS 25
#define LDS_STRIDE 40      // bf16 elems per LDS row (80B = 20 banks, conflict-free)

#if __has_builtin(__builtin_amdgcn_global_load_async_to_lds_b128) && \
    __has_builtin(__builtin_amdgcn_s_wait_asynccnt)
#define HAVE_ASYNC_LDS 1
// builtin signature (from hipcc diagnostic): param0 = int4 addrspace(1)*,
// param1 = int4 addrspace(3)*, then imm offset, imm cpol
typedef int b128vec_t __attribute__((vector_size(16)));
typedef __attribute__((address_space(1))) b128vec_t* gb128_t;
typedef __attribute__((address_space(3))) b128vec_t* lb128_t;
#else
#define HAVE_ASYNC_LDS 0
#endif

__device__ __forceinline__ void async_wait_all()
{
#if HAVE_ASYNC_LDS
    __builtin_amdgcn_s_wait_asynccnt(0);   // s_wait_asynccnt 0
#endif
}

// Stage one 256 x 32 bf16 kernel slice (row stride CINP) into an LDS tile
// (row stride LDS_STRIDE). 256 threads x 4 chunks of 16B.
template<int CINP>
__device__ __forceinline__ void stage_b_slice(const bf16_t* kg, bf16_t* dst, int tid)
{
#pragma unroll
    for (int r = 0; r < 4; ++r) {
        int i = tid + r * 256;
        int co = i >> 2, part = i & 3;
        const bf16_t* g = kg + (size_t)co * CINP + part * 8;
        bf16_t* l = dst + co * LDS_STRIDE + part * 8;
#if HAVE_ASYNC_LDS
        // generic LDS pointer's low 32 bits are the LDS byte offset
        __builtin_amdgcn_global_load_async_to_lds_b128(
            (gb128_t)(size_t)g, (lb128_t)(unsigned)(size_t)l, 0, 0);
#else
        *(uiv4*)l = *(const uiv4*)g;
#endif
    }
}

// ---------------------------------------------------------------------------
// 1) Convert x (B,100,700) f32 -> padded bf16 activation (B,136,704)
// ---------------------------------------------------------------------------
__global__ void convert_x_kernel(const float* __restrict__ x, bf16_t* __restrict__ xpad)
{
    size_t idx = (size_t)blockIdx.x * blockDim.x + threadIdx.x;
    const size_t total = (size_t)BATCH * TROWS * 704;
    if (idx >= total) return;
    int c = (int)(idx % 704);
    size_t r = idx / 704;
    int row = (int)(r % TROWS);
    int b   = (int)(r / TROWS);
    float v = 0.0f;
    if (row >= 24 && row < 24 + TREAL && c < 700)
        v = x[((size_t)b * TREAL + (row - 24)) * 700 + c];
    xpad[idx] = (bf16_t)v;
}

// ---------------------------------------------------------------------------
// 2) Dense DCLS kernel: Kb[tau][co][ci] = bf16( W * g(tau)/(sum g + eps) )
// ---------------------------------------------------------------------------
template<int CIN, int CINP, int COUT, int COUTP>
__global__ void build_dcls_kernel(const float* __restrict__ W, const float* __restrict__ P,
                                  bf16_t* __restrict__ kb)
{
    int idx = blockIdx.x * blockDim.x + threadIdx.x;
    if (idx >= COUTP * CINP) return;
    int ci = idx % CINP;
    int co = idx / CINP;
    float g[LTAPS] = {};
    float scale = 0.0f;
    bool valid = (co < COUT) && (ci < CIN);
    if (valid) {
        float p = P[co * CIN + ci];
        float s = 0.0f;
#pragma unroll
        for (int tau = 0; tau < LTAPS; ++tau) {
            float d = ((float)(tau - 12) - p) * (1.0f / 12.0f);
            g[tau] = expf(-0.5f * d * d);
            s += g[tau];
        }
        scale = W[co * CIN + ci] / (s + 1e-7f);
    }
#pragma unroll
    for (int tau = 0; tau < LTAPS; ++tau) {
        float v = valid ? g[tau] * scale : 0.0f;
        kb[((size_t)tau * COUTP + co) * CINP + ci] = (bf16_t)v;
    }
}

// ---------------------------------------------------------------------------
// 3) LDS-staged block conv for layers 1/2 (COUTP == 256).
//    Block = 1 batch, 256 thr = 8 waves. Wave w owns cout [32w, 32w+32),
//    all 7 time tiles -> 14 f32 accumulators. B is double-buffered: the
//    next tap's slice streams in (async) during the current tap's compute;
//    buffer parity makes a single end-of-tap barrier sufficient.
//    A frag (16x32 bf16): m=lane&15, hf=lane>>4: elems 0..7 K=hf*8..+7,
//    elems 8..15 K=hf*8+16..+23.  B frag (32x16): n=lane&15, same K order as A.
// ---------------------------------------------------------------------------
template<int CINP>
__global__ __launch_bounds__(256) void dcls_conv_lds(
    const bf16_t* __restrict__ xpad, const bf16_t* __restrict__ kb,
    float* __restrict__ y)
{
    constexpr int NCC = CINP / 32;
    __shared__ __align__(16) bf16_t Abuf[TROWS * LDS_STRIDE];      // 10.6 KB
    __shared__ __align__(16) bf16_t Bbuf[2][256 * LDS_STRIDE];     // 2 x 20 KB

    const int tid  = threadIdx.x;
    const int lane = tid & 31;
    const int wv   = tid >> 5;          // 0..7
    const int m    = lane & 15;
    const int hf   = lane >> 4;
    const int b    = blockIdx.x;        // batch
    const int co0  = wv * 32;           // two 16-wide cout tiles per wave

    const bf16_t* xb = xpad + (size_t)b * TROWS * CINP;

    v8f acc[14] = {};

    for (int cc = 0; cc < NCC; ++cc) {
        // ---- stage A: 136 rows x 32 ci (once per ci-chunk, reused by all taps)
        {
            const bf16_t* ag = xb + cc * 32;
            for (int i = tid; i < TROWS * 4; i += 256) {           // 16B chunks
                int row = i >> 2, part = i & 3;
                *(uiv4*)(Abuf + row * LDS_STRIDE + part * 8) =
                    *(const uiv4*)(ag + (size_t)row * CINP + part * 8);
            }
        }
        // ---- prime the B pipeline: tap 0 into buffer 0
        stage_b_slice<CINP>(kb + cc * 32, Bbuf[0], tid);
        async_wait_all();
        __syncthreads();

        for (int tau = 0; tau < LTAPS; ++tau) {
            const bf16_t* curB = Bbuf[tau & 1];
            // ---- stream next tap's slice into the other buffer (async DMA)
            if (tau + 1 < LTAPS)
                stage_b_slice<CINP>(kb + (size_t)(tau + 1) * 256 * CINP + cc * 32,
                                    Bbuf[(tau + 1) & 1], tid);

            // ---- B fragments for this wave's two cout tiles
            BFrag b0, b1;
            {
                const uiv4* bp0 = (const uiv4*)(curB + (co0 + m) * LDS_STRIDE + hf * 8);
                const uiv4* bp1 = (const uiv4*)(curB + (co0 + 16 + m) * LDS_STRIDE + hf * 8);
                b0.q[0] = bp0[0]; b0.q[1] = bp0[2];
                b1.q[0] = bp1[0]; b1.q[1] = bp1[2];
            }
            // ---- 7 time tiles x 2 cout tiles, A loads pipelined one tile ahead
            BFrag a0;
            {
                const uiv4* ap = (const uiv4*)(Abuf + (tau + m) * LDS_STRIDE + hf * 8);
                a0.q[0] = ap[0]; a0.q[1] = ap[2];
            }
#pragma unroll
            for (int mt = 0; mt < 7; ++mt) {
                BFrag an;
                if (mt < 6) {
                    const uiv4* ap = (const uiv4*)(Abuf +
                        ((mt + 1) * 16 + tau + m) * LDS_STRIDE + hf * 8);
                    an.q[0] = ap[0]; an.q[1] = ap[2];
                }
                acc[mt * 2 + 0] = __builtin_amdgcn_wmma_f32_16x16x32_bf16(
                    false, a0.v, false, b0.v, (short)0, acc[mt * 2 + 0], false, false);
                acc[mt * 2 + 1] = __builtin_amdgcn_wmma_f32_16x16x32_bf16(
                    false, a0.v, false, b1.v, (short)0, acc[mt * 2 + 1], false, false);
                if (mt < 6) a0 = an;
            }
            async_wait_all();   // our async writes into the other buffer done
            __syncthreads();    // all waves done reading curB / writing nxtB
        }
    }

    // ---- store: D elem r -> t = mt*16 + hf*8 + r, co = co0 + nt*16 + m
    float* yb = y + ((size_t)b * TPADO + hf * 8) * 256 + co0 + m;
#pragma unroll
    for (int mt = 0; mt < 7; ++mt)
#pragma unroll
        for (int nt = 0; nt < 2; ++nt)
#pragma unroll
            for (int r = 0; r < 8; ++r)
                yb[(size_t)(mt * 16 + r) * 256 + nt * 16] = acc[mt * 2 + nt][r];
}

// ---------------------------------------------------------------------------
// 3b) Simple per-wave WMMA conv for layer 3 (Cout padded to 32, ~2% of FLOPs)
// ---------------------------------------------------------------------------
template<int CINP, int COUTP, int NT>
__global__ __launch_bounds__(256) void dcls_conv_wmma(
    const bf16_t* __restrict__ xpad, const bf16_t* __restrict__ kb,
    float* __restrict__ y, int njobs)
{
    constexpr int NCHUNK = COUTP / (16 * NT);
    const int lane = threadIdx.x & 31;
    const int job  = blockIdx.x * (blockDim.x >> 5) + (threadIdx.x >> 5);
    if (job >= njobs) return;

    const int b    = job / (7 * NCHUNK);
    const int rem  = job % (7 * NCHUNK);
    const int t0   = (rem / NCHUNK) * 16;
    const int co0  = (rem % NCHUNK) * (16 * NT);

    const int m  = lane & 15;
    const int hf = lane >> 4;

    const bf16_t* abase = xpad + (size_t)b * TROWS * CINP + (size_t)(t0 + m) * CINP + hf * 8;
    const bf16_t* bbase = kb + (size_t)(co0 + m) * CINP + hf * 8;

    v8f acc[NT] = {};

    for (int tau = 0; tau < LTAPS; ++tau) {
        const bf16_t* arow = abase + (size_t)tau * CINP;
        const bf16_t* brow = bbase + (size_t)tau * COUTP * CINP;
        if (tau + 1 < LTAPS)
            __builtin_prefetch(brow + (size_t)COUTP * CINP, 0, 1);   // global_prefetch_b8
#pragma unroll 2
        for (int cc = 0; cc < CINP / 32; ++cc) {
            BFrag a;
            a.q[0] = *(const uiv4*)(arow + cc * 32);
            a.q[1] = *(const uiv4*)(arow + cc * 32 + 16);
#pragma unroll
            for (int nt = 0; nt < NT; ++nt) {
                BFrag bm;
                const bf16_t* bp = brow + (size_t)(nt * 16) * CINP + cc * 32;
                bm.q[0] = *(const uiv4*)(bp);
                bm.q[1] = *(const uiv4*)(bp + 16);
                acc[nt] = __builtin_amdgcn_wmma_f32_16x16x32_bf16(
                    false, a.v, false, bm.v, (short)0, acc[nt], false, false);
            }
        }
    }

    float* yb = y + ((size_t)b * TPADO + t0 + hf * 8) * COUTP + co0 + m;
#pragma unroll
    for (int nt = 0; nt < NT; ++nt)
#pragma unroll
        for (int r = 0; r < 8; ++r)
            yb[(size_t)r * COUTP + nt * 16] = acc[nt][r];
}

// ---------------------------------------------------------------------------
// 4) BN statistics over valid region t in [0,100): per-channel scale/shift
// ---------------------------------------------------------------------------
__global__ void bn_stats_kernel(const float* __restrict__ y, const float* __restrict__ gamma,
                                const float* __restrict__ beta, float* __restrict__ sc,
                                float* __restrict__ sh)
{
    const int c = blockIdx.x;          // 256 channels
    const int tid = threadIdx.x;       // 256 threads
    float s1 = 0.0f, s2 = 0.0f;
    for (int i = tid; i < BATCH * TREAL; i += 256) {
        int b = i / TREAL, t = i - b * TREAL;
        float v = y[((size_t)b * TPADO + t) * 256 + c];
        s1 += v; s2 += v * v;
    }
    __shared__ float r1[256], r2[256];
    r1[tid] = s1; r2[tid] = s2;
    __syncthreads();
    for (int off = 128; off > 0; off >>= 1) {
        if (tid < off) { r1[tid] += r1[tid + off]; r2[tid] += r2[tid + off]; }
        __syncthreads();
    }
    if (tid == 0) {
        const float N = (float)(BATCH * TREAL);
        float mean = r1[0] / N;
        float var  = r2[0] / N - mean * mean;
        float rstd = rsqrtf(var + 1e-5f);
        float g = gamma[c] * rstd;
        sc[c] = g;
        sh[c] = beta[c] - mean * g;
    }
}

// ---------------------------------------------------------------------------
// 5) BN + ReLU + bf16 convert into next padded activation buffer (B,136,256)
// ---------------------------------------------------------------------------
__global__ void bn_relu_pad_kernel(const float* __restrict__ y, const float* __restrict__ sc,
                                   const float* __restrict__ sh, bf16_t* __restrict__ hpad)
{
    size_t idx = (size_t)blockIdx.x * blockDim.x + threadIdx.x;
    const size_t total = (size_t)BATCH * TROWS * 256;
    if (idx >= total) return;
    int c = (int)(idx & 255);
    size_t r = idx >> 8;
    int row = (int)(r % TROWS);
    int b   = (int)(r / TROWS);
    float v = 0.0f;
    if (row >= 24) {
        int t = row - 24;                 // rows >=100 are junk, never read causally
        float xv = y[((size_t)b * TPADO + t) * 256 + c];
        xv = xv * sc[c] + sh[c];
        v = xv > 0.0f ? xv : 0.0f;
    }
    hpad[idx] = (bf16_t)v;
}

// ---------------------------------------------------------------------------
// 6) Per-step softmax over 20 classes, summed over t in [0,100) -> (B,20)
// ---------------------------------------------------------------------------
__global__ void softmax_sum_kernel(const float* __restrict__ y3, float* __restrict__ out)
{
    const int b = blockIdx.x;          // 128
    const int tid = threadIdx.x;       // 128 (t index)
    __shared__ float red[20][128];
    float p[20];
#pragma unroll
    for (int c = 0; c < 20; ++c) p[c] = 0.0f;
    if (tid < TREAL) {
        const float* row = y3 + ((size_t)b * TPADO + tid) * 32;
        float mx = -1e30f;
#pragma unroll
        for (int c = 0; c < 20; ++c) { p[c] = row[c]; mx = fmaxf(mx, p[c]); }
        float s = 0.0f;
#pragma unroll
        for (int c = 0; c < 20; ++c) { p[c] = expf(p[c] - mx); s += p[c]; }
        float inv = 1.0f / s;
#pragma unroll
        for (int c = 0; c < 20; ++c) p[c] *= inv;
    }
#pragma unroll
    for (int c = 0; c < 20; ++c) red[c][tid] = p[c];
    __syncthreads();
    for (int off = 64; off > 0; off >>= 1) {
        if (tid < off) {
#pragma unroll
            for (int c = 0; c < 20; ++c) red[c][tid] += red[c][tid + off];
        }
        __syncthreads();
    }
    if (tid < 20) out[b * 20 + tid] = red[tid][0];
}

// ---------------------------------------------------------------------------
// Host-side orchestration
// ---------------------------------------------------------------------------
extern "C" void kernel_launch(void* const* d_in, const int* in_sizes, int n_in,
                              void* d_out, int out_size, void* d_ws, size_t ws_size,
                              hipStream_t stream)
{
    const float* x  = (const float*)d_in[0];
    const float* W1 = (const float*)d_in[1];
    const float* P1 = (const float*)d_in[2];
    const float* W2 = (const float*)d_in[3];
    const float* P2 = (const float*)d_in[4];
    const float* W3 = (const float*)d_in[5];
    const float* P3 = (const float*)d_in[6];
    const float* g1 = (const float*)d_in[7];
    const float* b1 = (const float*)d_in[8];
    const float* g2 = (const float*)d_in[9];
    const float* b2 = (const float*)d_in[10];
    float* out = (float*)d_out;

    char* ws = (char*)d_ws;
    size_t off = 0;
    auto carve = [&](size_t bytes) -> void* {
        void* p = ws + off;
        off += (bytes + 255) & ~(size_t)255;
        return p;
    };
    bf16_t* xpad1 = (bf16_t*)carve((size_t)BATCH * TROWS * 704 * 2);   // 24.5 MB
    bf16_t* kb1   = (bf16_t*)carve((size_t)LTAPS * 256 * 704 * 2);     //  9.0 MB
    bf16_t* kb2   = (bf16_t*)carve((size_t)LTAPS * 256 * 256 * 2);     //  3.3 MB
    bf16_t* kb3   = (bf16_t*)carve((size_t)LTAPS * 32  * 256 * 2);     //  0.4 MB
    bf16_t* h1pad = (bf16_t*)carve((size_t)BATCH * TROWS * 256 * 2);   //  8.9 MB
    bf16_t* h2pad = (bf16_t*)carve((size_t)BATCH * TROWS * 256 * 2);   //  8.9 MB
    float*  ybuf  = (float*) carve((size_t)BATCH * TPADO * 256 * 4);   // 14.7 MB (reused 3x)
    float*  sc1   = (float*) carve(256 * 4);
    float*  sh1   = (float*) carve(256 * 4);
    float*  sc2   = (float*) carve(256 * 4);
    float*  sh2   = (float*) carve(256 * 4);

    // Stage inputs
    {
        size_t total = (size_t)BATCH * TROWS * 704;
        convert_x_kernel<<<(unsigned)((total + 255) / 256), 256, 0, stream>>>(x, xpad1);
    }
    build_dcls_kernel<700, 704, 256, 256><<<(256 * 704 + 255) / 256, 256, 0, stream>>>(W1, P1, kb1);
    build_dcls_kernel<256, 256, 256, 256><<<256, 256, 0, stream>>>(W2, P2, kb2);
    build_dcls_kernel<256, 256, 20, 32><<<32, 256, 0, stream>>>(W3, P3, kb3);

    // Layer 1: LDS-staged block GEMM with async double-buffered B
    dcls_conv_lds<704><<<BATCH, 256, 0, stream>>>(xpad1, kb1, ybuf);
    bn_stats_kernel<<<256, 256, 0, stream>>>(ybuf, g1, b1, sc1, sh1);
    {
        size_t total = (size_t)BATCH * TROWS * 256;
        bn_relu_pad_kernel<<<(unsigned)((total + 255) / 256), 256, 0, stream>>>(ybuf, sc1, sh1, h1pad);
    }

    // Layer 2
    dcls_conv_lds<256><<<BATCH, 256, 0, stream>>>(h1pad, kb2, ybuf);
    bn_stats_kernel<<<256, 256, 0, stream>>>(ybuf, g2, b2, sc2, sh2);
    {
        size_t total = (size_t)BATCH * TROWS * 256;
        bn_relu_pad_kernel<<<(unsigned)((total + 255) / 256), 256, 0, stream>>>(ybuf, sc2, sh2, h2pad);
    }

    // Layer 3: cout padded 20->32, NT=2 -> 128*7 = 896 wave-jobs
    dcls_conv_wmma<256, 32, 2><<<896 / 8, 256, 0, stream>>>(h2pad, kb3, ybuf, 896);

    // Softmax over classes, sum over time
    softmax_sum_kernel<<<BATCH, 128, 0, stream>>>(ybuf, out);
}
